// MultiHeadSelfAttention_3358664426216
// MI455X (gfx1250) — matrix operand order repaired and checked by reading, not compiled, not run on hardware
//
#include <hip/hip_runtime.h>
#include <hip/hip_bf16.h>

// ---------------------------------------------------------------------------
// Types for CDNA5 WMMA (wave32): v_wmma_f32_16x16x32_bf16
// ---------------------------------------------------------------------------
typedef __bf16 bf16_t;
typedef __attribute__((ext_vector_type(16))) __bf16 v16bf;
typedef __attribute__((ext_vector_type(8)))  __bf16 v8bf;
typedef __attribute__((ext_vector_type(8)))  float  v8f;

#define BATCH 2
#define SEQ   2048
#define DIM   2048
#define NH    16
#define NKV   4
#define HD    128
#define MROWS (BATCH * SEQ)   // 4096
#define KVW   (2 * NKV * HD)  // 1024

// ---------------------------------------------------------------------------
// Fragment loaders using contiguous k order for both A and B (not the ISA table k order)
// A (16x32 bf16, row-major source, leading dim lda):
//   lane = m + 16*half ; element e -> K = half*16 + e   (contiguous 16 bf16 per lane)
// B (32x16 bf16) loaded from an N-major ("transposed") source Bt[n][k]:
//   lane = n + 16*kh ; element e -> K = kh*16 + e   (contiguous 16 bf16 per lane)
// C/D (16x16 f32): VGPR v, lanes 0-15 -> M=v, N=lane ; lanes 16-31 -> M=v+8
// ---------------------------------------------------------------------------
__device__ __forceinline__ v16bf load_a_frag(const bf16_t* A, int lda) {
  const int lane = threadIdx.x & 31;
  const int m = lane & 15;
  const int h = lane >> 4;
  const bf16_t* row = A + m * lda;
  return *(const v16bf*)(row + h * 16);
}

__device__ __forceinline__ v16bf load_b_frag(const bf16_t* Bt, int ldb) {
  const int lane = threadIdx.x & 31;
  const int n  = lane & 15;
  const int kh = lane >> 4;
  return *(const v16bf*)(Bt + (size_t)n * ldb + kh * 16);
}

__device__ __forceinline__ float red_max16(float v) {
#pragma unroll
  for (int m = 1; m < 16; m <<= 1) v = fmaxf(v, __shfl_xor(v, m, 16));
  return v;
}
__device__ __forceinline__ float red_sum16(float v) {
#pragma unroll
  for (int m = 1; m < 16; m <<= 1) v += __shfl_xor(v, m, 16);
  return v;
}

// ---------------------------------------------------------------------------
// Prep kernels
// ---------------------------------------------------------------------------
__global__ void cvt_f32_bf16(const float* __restrict__ src, bf16_t* __restrict__ dst, int n) {
  int i = blockIdx.x * blockDim.x + threadIdx.x;
  if (i < n) dst[i] = (bf16_t)src[i];
}

// Wt[n][k] = W[k][n] ; W is K x N row-major fp32
__global__ void transpose_f32_bf16(const float* __restrict__ W, bf16_t* __restrict__ Wt,
                                   int K, int N) {
  int i = blockIdx.x * blockDim.x + threadIdx.x;
  if (i >= N * K) return;
  int nn = i / K;
  int kk = i - nn * K;
  Wt[(size_t)nn * K + kk] = (bf16_t)W[(size_t)kk * N + nn];
}

// Qraw [b][s][h*128] fp32 -> RoPE -> Qh [b][h][s][128] bf16
__global__ void rope_q_pack(const float* __restrict__ Qraw, const float* __restrict__ cosT,
                            const float* __restrict__ sinT, bf16_t* __restrict__ Qh) {
  int i = blockIdx.x * blockDim.x + threadIdx.x;   // B*S*NH*64 threads
  int j = i & 63;
  int h = (i >> 6) & (NH - 1);
  int s = (i >> 10) & (SEQ - 1);
  int b = i >> 21;
  const float* src = Qraw + (size_t)(b * SEQ + s) * DIM + h * HD;
  float x1 = src[j], x2 = src[64 + j];
  float c = cosT[s * 64 + j], sn = sinT[s * 64 + j];
  bf16_t* dst = Qh + ((size_t)(b * NH + h) * SEQ + s) * HD;
  dst[j]      = (bf16_t)(x1 * c - x2 * sn);
  dst[64 + j] = (bf16_t)(x1 * sn + x2 * c);
}

// KVraw [b][s][1024] fp32 -> K: RoPE -> Kh [b][hk][s][128] bf16
//                           V: transpose -> Vt [b][hk][128][s] bf16
__global__ void kv_pack(const float* __restrict__ KVraw, const float* __restrict__ cosT,
                        const float* __restrict__ sinT, bf16_t* __restrict__ Kh,
                        bf16_t* __restrict__ Vt) {
  int i = blockIdx.x * blockDim.x + threadIdx.x;   // B*S*NKV*64 threads
  int j  = i & 63;
  int hk = (i >> 6) & (NKV - 1);
  int s  = (i >> 8) & (SEQ - 1);
  int b  = i >> 19;
  const float* kv = KVraw + (size_t)(b * SEQ + s) * KVW;
  float k1 = kv[hk * HD + j],                 k2 = kv[hk * HD + 64 + j];
  float v1 = kv[NKV * HD + hk * HD + j],      v2 = kv[NKV * HD + hk * HD + 64 + j];
  float c = cosT[s * 64 + j], sn = sinT[s * 64 + j];
  bf16_t* kd = Kh + ((size_t)(b * NKV + hk) * SEQ + s) * HD;
  kd[j]      = (bf16_t)(k1 * c - k2 * sn);
  kd[64 + j] = (bf16_t)(k1 * sn + k2 * c);
  bf16_t* vd = Vt + (size_t)(b * NKV + hk) * HD * SEQ;
  vd[(size_t)j * SEQ + s]        = (bf16_t)v1;
  vd[(size_t)(64 + j) * SEQ + s] = (bf16_t)v2;
}

// ---------------------------------------------------------------------------
// Generic WMMA GEMM: C[M x N] = A[M x K](bf16,row) * Bt[N x K](bf16)^T + bias
// One wave owns a 32x64 C tile: 2 A-frags x 4 B-frags -> 8 wmma per 32-wide
// k-step (~21 FLOP/B of VMEM traffic). Prefetch next A k-chunk.
// ---------------------------------------------------------------------------
__global__ __launch_bounds__(32)
void gemm_bf16_wmma(const bf16_t* __restrict__ A, const bf16_t* __restrict__ Bt,
                    const float* __restrict__ bias, float* __restrict__ C,
                    int M, int N, int K) {
  const int n0 = blockIdx.x * 64;
  const int m0 = blockIdx.y * 32;
  v8f acc[2][4] = {};
  for (int k0 = 0; k0 < K; k0 += 32) {
    v16bf a0 = load_a_frag(A + (size_t)m0 * K + k0, K);
    v16bf a1 = load_a_frag(A + (size_t)(m0 + 16) * K + k0, K);
    if (k0 + 32 < K) {  // hint the next A chunk (global_prefetch_b8)
      __builtin_prefetch(A + (size_t)m0 * K + k0 + 32, 0, 1);
      __builtin_prefetch(A + (size_t)(m0 + 16) * K + k0 + 32, 0, 1);
    }
#pragma unroll
    for (int j = 0; j < 4; ++j) {
      v16bf b = load_b_frag(Bt + (size_t)(n0 + j * 16) * K + k0, K);
      acc[0][j] = __builtin_amdgcn_wmma_f32_16x16x32_bf16(false, a0, false, b,
                                                          (short)0, acc[0][j], false, false);
      acc[1][j] = __builtin_amdgcn_wmma_f32_16x16x32_bf16(false, a1, false, b,
                                                          (short)0, acc[1][j], false, false);
    }
  }
  const int lane = threadIdx.x & 31;
  const int n  = lane & 15;
  const int mo = (lane >> 4) * 8;
#pragma unroll
  for (int j = 0; j < 4; ++j) {
    float bv = bias[n0 + j * 16 + n];
#pragma unroll
    for (int i = 0; i < 2; ++i) {
#pragma unroll
      for (int v = 0; v < 8; ++v)
        C[(size_t)(m0 + i * 16 + v + mo) * N + n0 + j * 16 + n] = acc[i][j][v] + bv;
    }
  }
}

// ---------------------------------------------------------------------------
// Flash attention: one wave per (b, h, 16-query tile); 32-key blocks.
// S = Q K^T (8 wmma), online softmax (f32, half-wave reductions),
// P staged in LDS as bf16, O += P V (8 wmma). GQA: hk = h / 4.
// ---------------------------------------------------------------------------
__global__ __launch_bounds__(32)
void attn_wmma(const bf16_t* __restrict__ Qh, const bf16_t* __restrict__ Kh,
               const bf16_t* __restrict__ Vt, bf16_t* __restrict__ vo,
               const int* __restrict__ causal_p) {
  const int qt = blockIdx.x;
  const int h  = blockIdx.y;
  const int b  = blockIdx.z;
  const int q0 = qt * 16;
  const int hk = h / (NH / NKV);
  const bool causal = causal_p[0] != 0;
  const float scale = 0.08838834764831845f;  // 1/sqrt(128)

  const bf16_t* Qbase = Qh + ((size_t)(b * NH + h) * SEQ + q0) * HD;
  const bf16_t* Kbase = Kh + (size_t)(b * NKV + hk) * SEQ * HD;
  const bf16_t* Vbase = Vt + (size_t)(b * NKV + hk) * HD * SEQ;

  v16bf qf[4];
#pragma unroll
  for (int c = 0; c < 4; ++c) qf[c] = load_a_frag(Qbase + c * 32, HD);

  v8f oacc[8] = {};
  float mrow[8], lrow[8];
#pragma unroll
  for (int v = 0; v < 8; ++v) { mrow[v] = -3.0e38f; lrow[v] = 0.0f; }

  __shared__ __align__(32) bf16_t Pl[16 * 32];

  const int lane = threadIdx.x & 31;
  const int n  = lane & 15;
  const int mo = (lane >> 4) * 8;

  const int kend = causal ? (q0 + 16) : SEQ;
  for (int k0 = 0; k0 < kend; k0 += 32) {
    // S tiles for keys [k0, k0+16) and [k0+16, k0+32)
    v8f s0 = {}, s1 = {};
#pragma unroll
    for (int c = 0; c < 4; ++c) {
      v16bf bk0 = load_b_frag(Kbase + (size_t)k0 * HD + c * 32, HD);
      v16bf bk1 = load_b_frag(Kbase + (size_t)(k0 + 16) * HD + c * 32, HD);
      s0 = __builtin_amdgcn_wmma_f32_16x16x32_bf16(false, qf[c], false, bk0,
                                                   (short)0, s0, false, false);
      s1 = __builtin_amdgcn_wmma_f32_16x16x32_bf16(false, qf[c], false, bk1,
                                                   (short)0, s1, false, false);
    }
    // scale + causal mask + online softmax (rows live in 16-lane halves)
#pragma unroll
    for (int v = 0; v < 8; ++v) {
      const int qpos = q0 + v + mo;
      float a0 = s0[v] * scale;
      float a1 = s1[v] * scale;
      if (causal && (k0 + n      > qpos)) a0 = -1.0e30f;
      if (causal && (k0 + 16 + n > qpos)) a1 = -1.0e30f;
      float rm   = red_max16(fmaxf(a0, a1));
      float mnew = fmaxf(mrow[v], rm);
      float fac  = __expf(mrow[v] - mnew);
      float p0   = __expf(a0 - mnew);
      float p1   = __expf(a1 - mnew);
      lrow[v] = lrow[v] * fac + red_sum16(p0 + p1);
      mrow[v] = mnew;
#pragma unroll
      for (int t = 0; t < 8; ++t) oacc[t][v] *= fac;
      Pl[(v + mo) * 32 + n]      = (bf16_t)p0;
      Pl[(v + mo) * 32 + 16 + n] = (bf16_t)p1;
    }
    __syncthreads();
    v16bf pa = load_a_frag(Pl, 32);   // A fragment of P (16x32) via LDS
    __syncthreads();
#pragma unroll
    for (int t = 0; t < 8; ++t) {
      v16bf bv = load_b_frag(Vbase + (size_t)(t * 16) * SEQ + k0, SEQ);
      oacc[t] = __builtin_amdgcn_wmma_f32_16x16x32_bf16(false, pa, false, bv,
                                                        (short)0, oacc[t], false, false);
    }
  }

  // normalize and write vo[b][q][h*128 + d] (bf16, row-major for final GEMM)
#pragma unroll
  for (int t = 0; t < 8; ++t) {
#pragma unroll
    for (int v = 0; v < 8; ++v) {
      float val = oacc[t][v] / lrow[v];
      vo[(size_t)(b * SEQ + q0 + v + mo) * DIM + h * HD + t * 16 + n] = (bf16_t)val;
    }
  }
}

// ---------------------------------------------------------------------------
// Host-side launch
// ---------------------------------------------------------------------------
extern "C" void kernel_launch(void* const* d_in, const int* in_sizes, int n_in,
                              void* d_out, int out_size, void* d_ws, size_t ws_size,
                              hipStream_t stream) {
  (void)in_sizes; (void)n_in; (void)out_size; (void)ws_size;
  const float* x    = (const float*)d_in[0];
  const float* cosT = (const float*)d_in[1];
  const float* sinT = (const float*)d_in[2];
  const float* Wq   = (const float*)d_in[3];
  const float* bq   = (const float*)d_in[4];
  const float* Wkv  = (const float*)d_in[5];
  const float* bkv  = (const float*)d_in[6];
  const float* Wo   = (const float*)d_in[7];
  const float* bo   = (const float*)d_in[8];
  const int* causal = (const int*)d_in[9];
  float* out = (float*)d_out;

  char* ws = (char*)d_ws;
  size_t off = 0;
  auto carve = [&](size_t bytes) -> void* {
    void* p = ws + off;
    off = (off + bytes + 255) & ~(size_t)255;
    return p;
  };

  bf16_t* xb    = (bf16_t*)carve((size_t)MROWS * DIM * 2);            // 16 MB
  bf16_t* WqT   = (bf16_t*)carve((size_t)DIM * DIM * 2);              //  8 MB
  bf16_t* WkvT  = (bf16_t*)carve((size_t)KVW * DIM * 2);              //  4 MB
  bf16_t* WoT   = (bf16_t*)carve((size_t)DIM * DIM * 2);              //  8 MB
  float*  Qraw  = (float*) carve((size_t)MROWS * DIM * 4);            // 32 MB
  float*  KVraw = (float*) carve((size_t)MROWS * KVW * 4);            // 16 MB
  bf16_t* Qh    = (bf16_t*)carve((size_t)BATCH * NH * SEQ * HD * 2);  // 16 MB
  bf16_t* Kh    = (bf16_t*)carve((size_t)BATCH * NKV * SEQ * HD * 2); //  4 MB
  bf16_t* Vt    = (bf16_t*)carve((size_t)BATCH * NKV * HD * SEQ * 2); //  4 MB
  bf16_t* vo    = (bf16_t*)carve((size_t)MROWS * DIM * 2);            // 16 MB

  // 1) precision conversion + weight transposes
  { int n = MROWS * DIM;
    cvt_f32_bf16<<<(n + 255) / 256, 256, 0, stream>>>(x, xb, n); }
  { int n = DIM * DIM;
    transpose_f32_bf16<<<(n + 255) / 256, 256, 0, stream>>>(Wq, WqT, DIM, DIM); }
  { int n = DIM * KVW;
    transpose_f32_bf16<<<(n + 255) / 256, 256, 0, stream>>>(Wkv, WkvT, DIM, KVW); }
  { int n = DIM * DIM;
    transpose_f32_bf16<<<(n + 255) / 256, 256, 0, stream>>>(Wo, WoT, DIM, DIM); }

  // 2) QKV projections (WMMA)
  gemm_bf16_wmma<<<dim3(DIM / 64, MROWS / 32), 32, 0, stream>>>(
      xb, WqT, bq, Qraw, MROWS, DIM, DIM);
  gemm_bf16_wmma<<<dim3(KVW / 64, MROWS / 32), 32, 0, stream>>>(
      xb, WkvT, bkv, KVraw, MROWS, KVW, DIM);

  // 3) RoPE + head-major / transposed packing
  { int n = BATCH * SEQ * NH * 64;
    rope_q_pack<<<n / 256, 256, 0, stream>>>(Qraw, cosT, sinT, Qh); }
  { int n = BATCH * SEQ * NKV * 64;
    kv_pack<<<n / 256, 256, 0, stream>>>(KVraw, cosT, sinT, Kh, Vt); }

  // 4) flash attention (WMMA)
  attn_wmma<<<dim3(SEQ / 16, NH, BATCH), 32, 0, stream>>>(Qh, Kh, Vt, vo, causal);

  // 5) output projection (WMMA) -> fp32 out
  gemm_bf16_wmma<<<dim3(DIM / 64, MROWS / 32), 32, 0, stream>>>(
      vo, WoT, bo, out, MROWS, DIM, DIM);
}
